// Gemma2DecoderLayer_44272522887272
// MI455X (gfx1250) — matrix operand order repaired and checked by reading, not compiled, not run on hardware
//
#include <hip/hip_runtime.h>

typedef __bf16 bf16;
typedef __attribute__((ext_vector_type(16))) __bf16 v16bf;
typedef __attribute__((ext_vector_type(8)))  __bf16 v8bf;
typedef __attribute__((ext_vector_type(4)))  __bf16 v4bf;
typedef __attribute__((ext_vector_type(8)))  float  v8f;
typedef __attribute__((ext_vector_type(4)))  float  v4f;
typedef __attribute__((ext_vector_type(4)))  unsigned int u32x4;
typedef __attribute__((ext_vector_type(8)))  int i32x8;
typedef __attribute__((ext_vector_type(4)))  int i32x4;

constexpr int S_     = 4096;
constexpr int H_     = 2304;
constexpr int NQ_    = 8;
constexpr int NKV_   = 4;
constexpr int HD_    = 256;
constexpr int INTER_ = 9216;
constexpr int QKVW_  = (NQ_ + 2 * NKV_) * HD_;   // 4096
constexpr int OD_    = NQ_ * HD_;                // 2048
constexpr int WINDOW_ = 1024;
constexpr float SCALE_   = 0.0625f;              // 256^-0.5
constexpr float SOFTCAP_ = 50.0f;
constexpr float EPS_     = 1e-6f;

#if __has_builtin(__builtin_amdgcn_tensor_load_to_lds)
#define HAVE_TDM_ 1
#endif

__device__ inline v8f v8f_zero() {
  v8f r;
#pragma unroll
  for (int i = 0; i < 8; ++i) r[i] = 0.0f;
  return r;
}

__device__ inline v16bf make_frag(v8bf lo, v8bf hi) {
  v16bf r;
#pragma unroll
  for (int j = 0; j < 8; ++j) { r[j] = lo[j]; r[8 + j] = hi[j]; }
  return r;
}

__device__ inline v8f wmma_bf16(v16bf a, v16bf b, v8f c) {
  return __builtin_amdgcn_wmma_f32_16x16x32_bf16(false, a, false, b, (short)0, c,
                                                 false, false);
}

// ---- CDNA5 async global->LDS copy (GLOBAL_LOAD_ASYNC_TO_LDS_B128, ASYNCcnt)
__device__ inline void async_copy_b128(void* lds_dst, const void* gsrc) {
  uint32_t lo = (uint32_t)(uintptr_t)lds_dst;  // flat->LDS: addr[31:0]
  asm volatile("global_load_async_to_lds_b128 %0, %1, off"
               :
               : "v"(lo), "v"((uint64_t)(uintptr_t)gsrc)
               : "memory");
}

__device__ inline void wait_async() {
  asm volatile("s_wait_asynccnt 0" ::: "memory");
}

#ifdef HAVE_TDM_
// ---- CDNA5 Tensor Data Mover: 2D bf16 tile -> LDS with optional LDS padding.
// D# fields per CDNA5 ISA. Tensor dims == tile dims
// (view exactly covers the tile, no OOB). data_size = 2 bytes.
__device__ inline void tdm_load_tile_2d(void* lds_base, const void* gsrc,
                                        uint32_t tile_w, uint32_t tile_h,
                                        uint64_t row_stride_elems,
                                        uint32_t pad_en, uint32_t pad_interval,
                                        uint32_t pad_amount) {
  uint64_t ga = (uint64_t)(uintptr_t)gsrc;
  u32x4 g0;
  g0[0] = 1u;                                   // count=1 (valid user D#)
  g0[1] = (uint32_t)(uintptr_t)lds_base;        // lds_addr (bytes)
  g0[2] = (uint32_t)ga;                         // global_addr[31:0]
  g0[3] = (uint32_t)((ga >> 32) & 0x01FFFFFFu)  // global_addr[56:32]
          | 0x80000000u;                        // type=2 ("image")
  i32x8 g1;
  g1[0] = (int)((1u << 16)                      // data_size: 1 -> 2 bytes
                | (pad_en << 20) | (pad_interval << 22) | (pad_amount << 25));
  g1[1] = (int)((tile_w & 0xFFFFu) << 16);      // tensor_dim0[15:0] @ [63:48]
  g1[2] = (int)((tile_w >> 16) | ((tile_h & 0xFFFFu) << 16));  // d0 hi, d1 lo
  g1[3] = (int)((tile_h >> 16) | ((tile_w & 0xFFFFu) << 16));  // tile_dim0
  g1[4] = (int)(tile_h & 0xFFFFu);              // tile_dim1 (tile_dim2 = 0)
  g1[5] = (int)(uint32_t)row_stride_elems;      // tensor_dim0_stride[31:0]
  g1[6] = (int)(uint32_t)((row_stride_elems >> 32) & 0xFFFFu);  // stride hi16
  g1[7] = 0;
  i32x4 gz = {0, 0, 0, 0};
#if __clang_major__ >= 23
  i32x8 gz8 = {0, 0, 0, 0, 0, 0, 0, 0};
  __builtin_amdgcn_tensor_load_to_lds(g0, g1, gz, gz, gz8, 0);
#else
  __builtin_amdgcn_tensor_load_to_lds(g0, g1, gz, gz, 0);
#endif
}
#endif

__device__ inline float gelu_tanh(float x) {
  float x3 = x * x * x;
  float t = tanhf(0.7978845608028654f * (x + 0.044715f * x3));
  return 0.5f * x * (1.0f + t);
}

// ---------------------------------------------------------------- fp32->bf16
__global__ __launch_bounds__(256) void cvt_f32_bf16_kernel(
    const float* __restrict__ src, bf16* __restrict__ dst, long n4) {
  long i = (long)blockIdx.x * 256 + threadIdx.x;
  if (i >= n4) return;
  v4f x = *reinterpret_cast<const v4f*>(src + i * 4);
  v4bf y;
#pragma unroll
  for (int j = 0; j < 4; ++j) y[j] = (bf16)x[j];
  *reinterpret_cast<v4bf*>(dst + i * 4) = y;
}

// ---------------------------------------------------------------- rmsnorm
template <bool BF16OUT>
__global__ __launch_bounds__(256) void rmsnorm_kernel(
    const float* __restrict__ x, const float* __restrict__ w,
    void* __restrict__ out) {
  __shared__ float red[256];
  const int row = blockIdx.x, t = threadIdx.x;
  const float* xr = x + (size_t)row * H_;
  float xv[9];
  float ss = 0.f;
#pragma unroll
  for (int j = 0; j < 9; ++j) {
    xv[j] = xr[t + j * 256];
    ss += xv[j] * xv[j];
  }
  red[t] = ss;
  __syncthreads();
  for (int s = 128; s > 0; s >>= 1) {
    if (t < s) red[t] += red[t + s];
    __syncthreads();
  }
  float rs = rsqrtf(red[0] * (1.0f / H_) + EPS_);
#pragma unroll
  for (int j = 0; j < 9; ++j) {
    int i = t + j * 256;
    float v = xv[j] * rs * (1.0f + w[i]);
    if (BF16OUT)
      ((bf16*)out)[(size_t)row * H_ + i] = (bf16)v;
    else
      ((float*)out)[(size_t)row * H_ + i] = v;
  }
}

// -------------------------------------------- post-attn fused norm+residual
__global__ __launch_bounds__(256) void post_attn_kernel(
    const float* __restrict__ o, const float* __restrict__ hidden,
    const float* __restrict__ w_pa, const float* __restrict__ w_pf,
    float* __restrict__ resid, bf16* __restrict__ h2) {
  __shared__ float red[256];
  const int row = blockIdx.x, t = threadIdx.x;
  const float* orow = o + (size_t)row * H_;
  const float* hrow = hidden + (size_t)row * H_;
  float xv[9], rv[9];
  float ss = 0.f;
#pragma unroll
  for (int j = 0; j < 9; ++j) {
    xv[j] = orow[t + j * 256];
    ss += xv[j] * xv[j];
  }
  red[t] = ss;
  __syncthreads();
  for (int s = 128; s > 0; s >>= 1) {
    if (t < s) red[t] += red[t + s];
    __syncthreads();
  }
  float rs = rsqrtf(red[0] * (1.0f / H_) + EPS_);
  __syncthreads();
  float ss2 = 0.f;
#pragma unroll
  for (int j = 0; j < 9; ++j) {
    int i = t + j * 256;
    float r = xv[j] * rs * (1.0f + w_pa[i]) + hrow[i];
    rv[j] = r;
    ss2 += r * r;
    resid[(size_t)row * H_ + i] = r;
  }
  red[t] = ss2;
  __syncthreads();
  for (int s = 128; s > 0; s >>= 1) {
    if (t < s) red[t] += red[t + s];
    __syncthreads();
  }
  float rs2 = rsqrtf(red[0] * (1.0f / H_) + EPS_);
#pragma unroll
  for (int j = 0; j < 9; ++j) {
    int i = t + j * 256;
    h2[(size_t)row * H_ + i] = (bf16)(rv[j] * rs2 * (1.0f + w_pf[i]));
  }
}

// ---------------------------------------------------------------- RoPE/split
__global__ __launch_bounds__(128) void rope_split_kernel(
    const float* __restrict__ qkv, const int* __restrict__ pos,
    bf16* __restrict__ Qb, bf16* __restrict__ Kb, bf16* __restrict__ Vb) {
  const int s = blockIdx.x, y = blockIdx.y, i = threadIdx.x;  // i in 0..127
  const float p = (float)pos[s];
  const float* src;
  bf16* dst;
  bool do_rope;
  if (y < NQ_) {
    src = qkv + (size_t)s * QKVW_ + y * HD_;
    dst = Qb + ((size_t)y * S_ + s) * HD_;
    do_rope = true;
  } else if (y < NQ_ + NKV_) {
    int kh = y - NQ_;
    src = qkv + (size_t)s * QKVW_ + NQ_ * HD_ + kh * HD_;
    dst = Kb + ((size_t)kh * S_ + s) * HD_;
    do_rope = true;
  } else {
    int vh = y - NQ_ - NKV_;
    src = qkv + (size_t)s * QKVW_ + (NQ_ + NKV_) * HD_ + vh * HD_;
    dst = Vb + ((size_t)vh * S_ + s) * HD_;
    do_rope = false;
  }
  float x1 = src[i];
  float x2 = src[i + 128];
  if (do_rope) {
    float inv = __powf(10000.0f, -(float)i * (1.0f / 128.0f));
    float f = p * inv;
    float c = __cosf(f), sn = __sinf(f);
    dst[i] = (bf16)(x1 * c - x2 * sn);
    dst[i + 128] = (bf16)(x2 * c + x1 * sn);
  } else {
    dst[i] = (bf16)x1;
    dst[i + 128] = (bf16)x2;
  }
}

// ---------------------------------------------------------------- WMMA GEMM
// C[M,N](f32) = A[M,K](bf16,row-major) * B[K,N](bf16,row-major)
// EPI==0: store f32 C.  EPI==1: obf = (bf16)(gelu_tanh(C) * aux[C-index])
template <int EPI>
__global__ __launch_bounds__(256) void gemm_bf16_kernel(
    const bf16* __restrict__ A, const bf16* __restrict__ B,
    float* __restrict__ C, const float* __restrict__ aux,
    bf16* __restrict__ obf, int K, int lda, int ldb, int ldc) {
  __shared__ __align__(16) bf16 Alds[128 * 40];
  __shared__ __align__(16) bf16 BT[128 * 40];

  const int tid = threadIdx.x;
  const int lane = tid & 31;
  const int wid = tid >> 5;
  const int lhi = lane >> 4;   // 0/1 lane half
  const int llo = lane & 15;
  const int bm = blockIdx.y * 128;
  const int bn = blockIdx.x * 128;
  const int wm = (wid >> 1) * 32;  // wave M offset (4 wave-rows)
  const int wn = (wid & 1) * 64;   // wave N offset (2 wave-cols)

  v8f acc[2][4];
#pragma unroll
  for (int mi = 0; mi < 2; ++mi)
#pragma unroll
    for (int ni = 0; ni < 4; ++ni) acc[mi][ni] = v8f_zero();

  for (int kb = 0; kb < K; kb += 32) {
    __syncthreads();
#ifdef HAVE_TDM_
    // stage A tile 128x32 via TDM: one tensor_load_to_lds per K-step.
    // LDS padding: 64B rows + 16B pad -> stride 40 halves (pad_int=3, amt=3).
    if (wid == 0)
      tdm_load_tile_2d(Alds, &A[(size_t)bm * lda + kb], /*w*/ 32, /*h*/ 128,
                       (uint64_t)lda, /*pad*/ 1, 3, 3);
#else
    // fallback: per-lane async DMA -> LDS
#pragma unroll
    for (int u = tid; u < 512; u += 256) {
      int row = u >> 2, off = (u & 3) * 8;
      async_copy_b128(&Alds[row * 40 + off],
                      &A[(size_t)(bm + row) * lda + kb + off]);
    }
#endif
    // stage B tile 32x128 transposed -> BT[n][k] stride 40 (manual transpose)
#pragma unroll
    for (int u = tid; u < 512; u += 256) {
      int kr = u >> 4, off = (u & 15) * 8;
      v8bf b8 = *reinterpret_cast<const v8bf*>(
          &B[(size_t)(kb + kr) * ldb + bn + off]);
#pragma unroll
      for (int j = 0; j < 8; ++j) BT[(off + j) * 40 + kr] = b8[j];
    }
    if (kb + 32 < K) {  // hint next tiles toward L2 (global_prefetch_b8)
      __builtin_prefetch(&A[(size_t)(bm + (tid >> 1)) * lda + kb + 32], 0, 1);
      __builtin_prefetch(&B[(size_t)(kb + 32 + (tid >> 3)) * ldb + bn], 0, 1);
    }
#ifdef HAVE_TDM_
    __builtin_amdgcn_s_wait_tensorcnt(0);  // TENSORcnt==0 before barrier
#else
    wait_async();
#endif
    __syncthreads();

    v16bf af[2], bfv[4];
#pragma unroll
    for (int mi = 0; mi < 2; ++mi) {
      const bf16* p = &Alds[(wm + mi * 16 + llo) * 40 + lhi * 8];
      af[mi] = make_frag(*reinterpret_cast<const v8bf*>(p),
                         *reinterpret_cast<const v8bf*>(p + 16));
    }
#pragma unroll
    for (int ni = 0; ni < 4; ++ni) {
      const bf16* p = &BT[(wn + ni * 16 + llo) * 40 + lhi * 8];
      bfv[ni] = make_frag(*reinterpret_cast<const v8bf*>(p),
                          *reinterpret_cast<const v8bf*>(p + 16));
    }
#pragma unroll
    for (int mi = 0; mi < 2; ++mi)
#pragma unroll
      for (int ni = 0; ni < 4; ++ni)
        acc[mi][ni] = wmma_bf16(af[mi], bfv[ni], acc[mi][ni]);
  }

  // epilogue: C layout -> lane=col(llo), vgpr v = row (lhi*8+v)
#pragma unroll
  for (int mi = 0; mi < 2; ++mi) {
#pragma unroll
    for (int ni = 0; ni < 4; ++ni) {
#pragma unroll
      for (int v = 0; v < 8; ++v) {
        int row = bm + wm + mi * 16 + lhi * 8 + v;
        int col = bn + wn + ni * 16 + llo;
        size_t idx = (size_t)row * ldc + col;
        if (EPI == 0) {
          C[idx] = acc[mi][ni][v];
        } else {
          float g = gelu_tanh(acc[mi][ni][v]);
          obf[idx] = (bf16)(g * aux[idx]);
        }
      }
    }
  }
}

// ------------------------------------------------------------- attention
// grid: (S/64, NQ). block = 128 (4 waves). wave w handles 16 queries.
__global__ __launch_bounds__(128) void attn_kernel(
    const bf16* __restrict__ Q, const bf16* __restrict__ Kk,
    const bf16* __restrict__ V, bf16* __restrict__ Out) {
  __shared__ __align__(16) bf16 Klds[32 * 256];
  __shared__ __align__(16) bf16 VT[256 * 40];
  __shared__ __align__(16) bf16 Plds[4][16 * 40];

  const int qblk = blockIdx.x * 64;
  const int h = blockIdx.y;
  const int kvh = h >> 1;  // NQ/NKV == 2
  const int tid = threadIdx.x;
  const int wid = tid >> 5;
  const int lane = tid & 31;
  const int lhi = lane >> 4;
  const int llo = lane & 15;
  const int qbase = qblk + wid * 16;

  // Q fragments for this wave's 16 rows (A layout), held in registers
  v16bf qf[8];
  {
    const bf16* qrow = Q + ((size_t)h * S_ + (qbase + llo)) * HD_;
#pragma unroll
    for (int c = 0; c < 8; ++c) {
      int d0 = c * 32 + lhi * 8;
      qf[c] = make_frag(*reinterpret_cast<const v8bf*>(qrow + d0),
                        *reinterpret_cast<const v8bf*>(qrow + d0 + 16));
    }
  }

  v8f o[16];
#pragma unroll
  for (int t = 0; t < 16; ++t) o[t] = v8f_zero();
  float mrun[8], lrun[8];
#pragma unroll
  for (int v = 0; v < 8; ++v) {
    mrun[v] = -3.0e38f;
    lrun[v] = 0.f;
  }

  int klo = qblk - (WINDOW_ - 1);
  if (klo < 0) klo = 0;
  klo &= ~31;
  const int khi = qblk + 64;

  for (int kb = klo; kb < khi; kb += 32) {
    __syncthreads();
    // stage K tile [32][256] via per-lane async DMA; V transposed manually
    for (int u = tid; u < 1024; u += 128) {
      int krow = u >> 5, off = (u & 31) * 8;
      const size_t gofs = ((size_t)kvh * S_ + kb + krow) * HD_ + off;
      async_copy_b128(&Klds[krow * 256 + off], Kk + gofs);
      v8bf v8 = *reinterpret_cast<const v8bf*>(V + gofs);
#pragma unroll
      for (int j = 0; j < 8; ++j) VT[(off + j) * 40 + krow] = v8[j];
    }
    wait_async();
    __syncthreads();

    // scores for 2 key tiles of 16
    v8f sc[2];
    sc[0] = v8f_zero();
    sc[1] = v8f_zero();
#pragma unroll
    for (int t = 0; t < 2; ++t) {
#pragma unroll
      for (int c = 0; c < 8; ++c) {
        const bf16* kp = &Klds[(t * 16 + llo) * 256 + c * 32 + lhi * 8];
        v16bf bfr = make_frag(*reinterpret_cast<const v8bf*>(kp),
                              *reinterpret_cast<const v8bf*>(kp + 16));
        sc[t] = wmma_bf16(qf[c], bfr, sc[t]);
      }
    }

    // softcap + mask + online softmax (row stats per lane-half, 8 rows each)
    float p0[8], p1[8];
#pragma unroll
    for (int v = 0; v < 8; ++v) {
      int qi = qbase + lhi * 8 + v;
      int k0 = kb + llo, k1 = kb + 16 + llo;
      float x0 = SOFTCAP_ * tanhf(sc[0][v] * (SCALE_ / SOFTCAP_));
      float x1 = SOFTCAP_ * tanhf(sc[1][v] * (SCALE_ / SOFTCAP_));
      bool ok0 = (k0 <= qi) && (qi - k0 < WINDOW_);
      bool ok1 = (k1 <= qi) && (qi - k1 < WINDOW_);
      float y0 = ok0 ? x0 : -3.0e38f;
      float y1 = ok1 ? x1 : -3.0e38f;
      float hm = fmaxf(y0, y1);
#pragma unroll
      for (int sh = 1; sh < 16; sh <<= 1) hm = fmaxf(hm, __shfl_xor(hm, sh, 32));
      float newm = fmaxf(mrun[v], hm);
      float corr = __expf(mrun[v] - newm);
      p0[v] = ok0 ? __expf(x0 - newm) : 0.f;
      p1[v] = ok1 ? __expf(x1 - newm) : 0.f;
      float ps = p0[v] + p1[v];
#pragma unroll
      for (int sh = 1; sh < 16; sh <<= 1) ps += __shfl_xor(ps, sh, 32);
      lrun[v] = lrun[v] * corr + ps;
      mrun[v] = newm;
#pragma unroll
      for (int t = 0; t < 16; ++t) o[t][v] *= corr;
    }

    // redistribute P (C layout) -> A layout via wave-private LDS
    bf16* pb = &Plds[wid][0];
#pragma unroll
    for (int v = 0; v < 8; ++v) {
      int m = lhi * 8 + v;
      pb[m * 40 + llo] = (bf16)p0[v];
      pb[m * 40 + 16 + llo] = (bf16)p1[v];
    }
    v16bf pf;
    {
      const bf16* pr = pb + llo * 40 + lhi * 8;
      pf = make_frag(*reinterpret_cast<const v8bf*>(pr),
                     *reinterpret_cast<const v8bf*>(pr + 16));
    }

    // O += P @ V  (16 n-tiles over HD=256)
#pragma unroll
    for (int nt = 0; nt < 16; ++nt) {
      const bf16* vp = &VT[(nt * 16 + llo) * 40 + lhi * 8];
      v16bf bfr = make_frag(*reinterpret_cast<const v8bf*>(vp),
                            *reinterpret_cast<const v8bf*>(vp + 16));
      o[nt] = wmma_bf16(pf, bfr, o[nt]);
    }
  }

  // epilogue: normalize and scatter to [S][NQ*HD]
#pragma unroll
  for (int v = 0; v < 8; ++v) {
    float inv = 1.0f / lrun[v];
    int qi = qbase + lhi * 8 + v;
    bf16* orow = Out + (size_t)qi * OD_ + h * HD_;
#pragma unroll
    for (int nt = 0; nt < 16; ++nt)
      orow[nt * 16 + llo] = (bf16)(o[nt][v] * inv);
  }
}

// ---------------------------------------------------------------- launcher
extern "C" void kernel_launch(void* const* d_in, const int* in_sizes, int n_in,
                              void* d_out, int out_size, void* d_ws,
                              size_t ws_size, hipStream_t stream) {
  const int* positions = (const int*)d_in[0];
  const float* hidden = (const float*)d_in[1];
  const float* w_qkv = (const float*)d_in[2];
  const float* w_o = (const float*)d_in[3];
  const float* w_gu = (const float*)d_in[4];
  const float* w_dn = (const float*)d_in[5];
  const float* w_in_ln = (const float*)d_in[6];
  const float* w_pa_ln = (const float*)d_in[7];
  const float* w_pf_ln = (const float*)d_in[8];
  const float* w_pff_ln = (const float*)d_in[9];
  float* out = (float*)d_out;                // h
  float* resid_out = out + (size_t)S_ * H_;  // residual

  size_t off = 0;
  auto carve = [&](size_t bytes) -> void* {
    void* p = (char*)d_ws + off;
    off = (off + bytes + 255) & ~(size_t)255;
    return p;
  };
  bf16* wqkv_bf = (bf16*)carve((size_t)H_ * QKVW_ * 2);
  bf16* wo_bf = (bf16*)carve((size_t)OD_ * H_ * 2);
  bf16* wgu_bf = (bf16*)carve((size_t)H_ * 2 * INTER_ * 2);
  bf16* wdn_bf = (bf16*)carve((size_t)INTER_ * H_ * 2);
  bf16* hnorm = (bf16*)carve((size_t)S_ * H_ * 2);
  float* qkv_f = (float*)carve((size_t)S_ * QKVW_ * 4);
  bf16* Qb = (bf16*)carve((size_t)NQ_ * S_ * HD_ * 2);
  bf16* Kb = (bf16*)carve((size_t)NKV_ * S_ * HD_ * 2);
  bf16* Vb = (bf16*)carve((size_t)NKV_ * S_ * HD_ * 2);
  bf16* attn_bf = (bf16*)carve((size_t)S_ * OD_ * 2);
  float* o_f = (float*)carve((size_t)S_ * H_ * 4);
  bf16* h2 = (bf16*)carve((size_t)S_ * H_ * 2);
  float* up_f = (float*)carve((size_t)S_ * INTER_ * 4);
  bf16* act_bf = (bf16*)carve((size_t)S_ * INTER_ * 2);
  float* mlp_f = (float*)carve((size_t)S_ * H_ * 4);
  (void)ws_size; (void)n_in; (void)in_sizes; (void)out_size;

  auto cvt = [&](const float* s, bf16* d, size_t n) {
    long n4 = (long)(n / 4);
    cvt_f32_bf16_kernel<<<(unsigned)((n4 + 255) / 256), 256, 0, stream>>>(s, d, n4);
  };
  cvt(w_qkv, wqkv_bf, (size_t)H_ * QKVW_);
  cvt(w_o, wo_bf, (size_t)OD_ * H_);
  cvt(w_gu, wgu_bf, (size_t)H_ * 2 * INTER_);
  cvt(w_dn, wdn_bf, (size_t)INTER_ * H_);

  // input rmsnorm -> bf16
  rmsnorm_kernel<true><<<S_, 256, 0, stream>>>(hidden, w_in_ln, (void*)hnorm);

  // QKV projection: [S,H] x [H,QKVW]
  gemm_bf16_kernel<0><<<dim3(QKVW_ / 128, S_ / 128), 256, 0, stream>>>(
      hnorm, wqkv_bf, qkv_f, nullptr, nullptr, H_, H_, QKVW_, QKVW_);

  // RoPE + head split
  rope_split_kernel<<<dim3(S_, NQ_ + 2 * NKV_), 128, 0, stream>>>(
      qkv_f, positions, Qb, Kb, Vb);

  // sliding-window flash attention
  attn_kernel<<<dim3(S_ / 64, NQ_), 128, 0, stream>>>(Qb, Kb, Vb, attn_bf);

  // O projection: [S,OD] x [OD,H]
  gemm_bf16_kernel<0><<<dim3(H_ / 128, S_ / 128), 256, 0, stream>>>(
      attn_bf, wo_bf, o_f, nullptr, nullptr, OD_, OD_, H_, H_);

  // post-attn rmsnorm + residual -> resid_out ; pre-ff rmsnorm -> h2(bf16)
  post_attn_kernel<<<S_, 256, 0, stream>>>(o_f, hidden, w_pa_ln, w_pf_ln,
                                           resid_out, h2);

  // up = h2 @ W_up  (columns INTER..2*INTER of w_gate_up)
  gemm_bf16_kernel<0><<<dim3(INTER_ / 128, S_ / 128), 256, 0, stream>>>(
      h2, wgu_bf + INTER_, up_f, nullptr, nullptr, H_, H_, 2 * INTER_, INTER_);

  // act = gelu(h2 @ W_gate) * up  -> bf16
  gemm_bf16_kernel<1><<<dim3(INTER_ / 128, S_ / 128), 256, 0, stream>>>(
      h2, wgu_bf, nullptr, up_f, act_bf, H_, H_, 2 * INTER_, INTER_);

  // down: [S,INTER] x [INTER,H]
  gemm_bf16_kernel<0><<<dim3(H_ / 128, S_ / 128), 256, 0, stream>>>(
      act_bf, wdn_bf, mlp_f, nullptr, nullptr, INTER_, INTER_, H_, H_);

  // final rmsnorm -> h (f32)
  rmsnorm_kernel<false><<<S_, 256, 0, stream>>>(mlp_f, w_pff_ln, (void*)out);
}
